// SelfAttentionFCLayer_29463475651293
// MI455X (gfx1250) — matrix operand order repaired and checked by reading, not compiled, not run on hardware
//
#include <hip/hip_runtime.h>

typedef __attribute__((ext_vector_type(8)))  float  v8f;
typedef __attribute__((ext_vector_type(16))) __bf16 v16bf;
typedef __attribute__((ext_vector_type(8)))  __bf16 v8bf;

constexpr int BB = 32;
constexpr int NN = 1024;
constexpr int CC = 64;
constexpr int GG = 64;

__device__ __forceinline__ v16bf mk16(v8bf lo, v8bf hi) {
    return __builtin_shufflevector(lo, hi, 0,1,2,3,4,5,6,7,8,9,10,11,12,13,14,15);
}
__device__ __forceinline__ v8bf cvt8(v8f f) {
    return __builtin_convertvector(f, v8bf);   // native v_cvt_pk_bf16_f32
}

// ---- Kernel 0: W[c][g] (f32) -> Wt[g][c] (bf16), 3 matrices -------------
__global__ void sattn_wt_kernel(const float* __restrict__ Wq,
                                const float* __restrict__ Wk,
                                const float* __restrict__ Wv,
                                __bf16* __restrict__ wt) {
    const float* W = (blockIdx.x == 0) ? Wq : (blockIdx.x == 1 ? Wk : Wv);
    __bf16* o = wt + blockIdx.x * (CC * GG);
    for (int i = threadIdx.x; i < CC * GG; i += blockDim.x) {
        int g = i >> 6, c = i & 63;
        o[i] = (__bf16)W[c * GG + g];            // o[g*64 + c]
    }
}

// ---- Kernel 1: QKV projection via bf16 WMMA ------------------------------
// One wave per (batch, 16-row tile). Q,K row-major bf16; V stored transposed.
__global__ void __launch_bounds__(32)
sattn_qkv_kernel(const float* __restrict__ x,
                 const __bf16* __restrict__ wt,
                 const float* __restrict__ bq,
                 const float* __restrict__ bk,
                 const float* __restrict__ bv,
                 __bf16* __restrict__ Qb,
                 __bf16* __restrict__ Kb,
                 __bf16* __restrict__ Vt) {
    const int b    = blockIdx.x >> 6;
    const int qt   = blockIdx.x & 63;
    const int lane = threadIdx.x & 31;
    const int col  = lane & 15;
    const int hi   = lane >> 4;

    // A fragments: x rows qt*16..+16, two K-chunks of 32 (C=64).
    // 16-bit A layout: lanes0-15 K=c*32+0..15; lanes16-31 K=c*32+16..31 (contiguous).
    const float* xr = x + (size_t)(b * NN + qt * 16 + col) * CC;
    v16bf aF[2];
#pragma unroll
    for (int c = 0; c < 2; ++c) {
        v8f f0 = *(const v8f*)(xr + c * 32 + hi * 16);
        v8f f1 = *(const v8f*)(xr + c * 32 + hi * 16 + 8);
        aF[c] = mk16(cvt8(f0), cvt8(f1));
    }

    const float* biases[3] = { bq, bk, bv };
#pragma unroll
    for (int p = 0; p < 3; ++p) {
        const __bf16* Wtp = wt + p * (CC * GG);
#pragma unroll
        for (int j = 0; j < 4; ++j) {
            const int g = j * 16 + col;              // output column
            const float bias = biases[p][g];
            v8f acc;
#pragma unroll
            for (int r = 0; r < 8; ++r) acc[r] = bias;   // bias rides in C
#pragma unroll
            for (int c = 0; c < 2; ++c) {
                // B fragment: lane holds col g, K rows c*32 + hi*16 .. +16 (contiguous)
                v16bf bF = *(const v16bf*)(Wtp + g * CC + c * 32 + hi * 16);
                acc = __builtin_amdgcn_wmma_f32_16x16x32_bf16(
                          false, aF[c], false, bF, (short)0, acc, false, false);
            }
            v8bf ab = cvt8(acc);
            if (p == 2) {
                // V transposed: Vt[b][g][n]; lane's 8 rows are keys hi*8..+8 (contiguous)
                *(v8bf*)(Vt + (size_t)(b * GG + g) * NN + qt * 16 + hi * 8) = ab;
            } else {
                __bf16* dst = (p == 0) ? Qb : Kb;
#pragma unroll
                for (int r = 0; r < 8; ++r)
                    dst[(size_t)(b * NN + qt * 16 + hi * 8 + r) * GG + g] = ab[r];
            }
        }
    }
}

// ---- Kernel 2: streaming (flash-style) masked attention ------------------
__global__ void __launch_bounds__(32)
sattn_attn_kernel(const __bf16* __restrict__ Qb,
                  const __bf16* __restrict__ Kb,
                  const __bf16* __restrict__ Vt,
                  const int* __restrict__ valid_len,
                  float* __restrict__ out) {
    __shared__ __bf16 pTile[16 * 32];   // P tile, row-major [m][k]
    const int b    = blockIdx.x >> 6;
    const int qt   = blockIdx.x & 63;
    const int lane = threadIdx.x & 31;
    const int col  = lane & 15;
    const int hi   = lane >> 4;
    const int vl   = valid_len[b];
    const float L2E = 1.44269504088896340736f;

    // Masked key blocks beyond valid_len contribute exactly zero (f32 exp
    // underflow) once any valid key has been seen; block 0 always has a valid
    // key when vl > 0, so we can stop at ceil(vl/32). vl == 0 needs the full
    // loop (reference degenerates to the mean of V).
    const int kend = (vl == 0) ? NN : ((vl + 31) & ~31);

    // Q fragments (A layout) from row-major bf16 Q
    const __bf16* qr = Qb + (size_t)(b * NN + qt * 16 + col) * GG;
    v16bf qF[2];
#pragma unroll
    for (int c = 0; c < 2; ++c) {
        v8bf lo = *(const v8bf*)(qr + c * 32 + hi * 16);
        v8bf h8 = *(const v8bf*)(qr + c * 32 + hi * 16 + 8);
        qF[c] = mk16(lo, h8);
    }

    // All-ones B fragment: P @ 1 computes per-row sums on the matrix pipe
    v16bf onesF;
#pragma unroll
    for (int e = 0; e < 16; ++e) onesF[e] = (__bf16)1.0f;

    // Loop-invariant zero C operand for the first WMMA of each score chain
    v8f zc;
#pragma unroll
    for (int r = 0; r < 8; ++r) zc[r] = 0.0f;

    float m_r[8];
    v8f   lvec;      // per-row running denominators (same D-layout rows as O)
    v8f   O[4];
#pragma unroll
    for (int r = 0; r < 8; ++r) { m_r[r] = -3.0e38f; lvec[r] = 0.0f; }
#pragma unroll
    for (int j = 0; j < 4; ++j)
#pragma unroll
        for (int r = 0; r < 8; ++r) O[j][r] = 0.0f;

    for (int kb = 0; kb < kend; kb += 32) {
        // prefetch next key-block's K and V rows (global_prefetch_b8)
        if (kb + 32 < kend) {
            __builtin_prefetch(Kb + (size_t)(b * NN + kb + 32 + col) * GG, 0, 0);
            __builtin_prefetch(Kb + (size_t)(b * NN + kb + 48 + col) * GG, 0, 0);
            __builtin_prefetch(Vt + (size_t)(b * GG + col) * NN + kb + 32, 0, 0);
            __builtin_prefetch(Vt + (size_t)(b * GG + 48 + col) * NN + kb + 32, 0, 0);
        }
        // ---- scores for 32 keys: two 16x16 D tiles, K=64 in two chunks ----
        const __bf16* kr0 = Kb + (size_t)(b * NN + kb + col) * GG;
        const __bf16* kr1 = Kb + (size_t)(b * NN + kb + 16 + col) * GG;
        v16bf k00 = *(const v16bf*)(kr0 + hi * 16);
        v16bf k01 = *(const v16bf*)(kr0 + 32 + hi * 16);
        v16bf k10 = *(const v16bf*)(kr1 + hi * 16);
        v16bf k11 = *(const v16bf*)(kr1 + 32 + hi * 16);
        v8f s0 = __builtin_amdgcn_wmma_f32_16x16x32_bf16(false, qF[0], false, k00, (short)0, zc, false, false);
        s0     = __builtin_amdgcn_wmma_f32_16x16x32_bf16(false, qF[1], false, k01, (short)0, s0, false, false);
        v8f s1 = __builtin_amdgcn_wmma_f32_16x16x32_bf16(false, qF[0], false, k10, (short)0, zc, false, false);
        s1     = __builtin_amdgcn_wmma_f32_16x16x32_bf16(false, qF[1], false, k11, (short)0, s1, false, false);

        // ---- key mask (selects only; EXEC stays full for WMMA) ----
        const bool msk0 = (kb + col)      >= vl;
        const bool msk1 = (kb + 16 + col) >= vl;
#pragma unroll
        for (int r = 0; r < 8; ++r) {
            s0[r] = msk0 ? -1000000.0f : s0[r];
            s1[r] = msk1 ? -1000000.0f : s1[r];
        }
        // ---- online softmax: row max via 4-step butterfly (cols 0..15) ----
#pragma unroll
        for (int r = 0; r < 8; ++r) {
            float t = fmaxf(s0[r], s1[r]);
            t = fmaxf(t, __shfl_xor(t, 1, 32));
            t = fmaxf(t, __shfl_xor(t, 2, 32));
            t = fmaxf(t, __shfl_xor(t, 4, 32));
            t = fmaxf(t, __shfl_xor(t, 8, 32));
            const float mnew  = fmaxf(m_r[r], t);
            const float scale = __builtin_amdgcn_exp2f((m_r[r] - mnew) * L2E);
            m_r[r] = mnew;
            s0[r] = __builtin_amdgcn_exp2f((s0[r] - mnew) * L2E);
            s1[r] = __builtin_amdgcn_exp2f((s1[r] - mnew) * L2E);
            lvec[r] *= scale;
#pragma unroll
            for (int j = 0; j < 4; ++j) O[j][r] *= scale;
        }

        // ---- D-layout -> A-layout via LDS, P in bf16 ----
        v8bf p0b = cvt8(s0), p1b = cvt8(s1);
#pragma unroll
        for (int r = 0; r < 8; ++r) {
            pTile[(hi * 8 + r) * 32 + col]      = p0b[r];
            pTile[(hi * 8 + r) * 32 + 16 + col] = p1b[r];
        }
        __syncthreads();
        {
            v8bf plo = *(const v8bf*)(pTile + col * 32 + hi * 16);
            v8bf phi = *(const v8bf*)(pTile + col * 32 + hi * 16 + 8);
            v16bf pF = mk16(plo, phi);
            // row-sum on the matrix pipe: l += P @ ones
            lvec = __builtin_amdgcn_wmma_f32_16x16x32_bf16(false, pF, false, onesF, (short)0, lvec, false, false);
#pragma unroll
            for (int j = 0; j < 4; ++j) {
                v16bf vF = *(const v16bf*)(Vt + (size_t)(b * GG + j * 16 + col) * NN + kb + hi * 16);
                O[j] = __builtin_amdgcn_wmma_f32_16x16x32_bf16(false, pF, false, vF, (short)0, O[j], false, false);
            }
        }
        __syncthreads();
    }

    // ---- normalize (v_rcp_f32 + muls) and store [B,N,G] f32 ----
    float rinv[8];
#pragma unroll
    for (int r = 0; r < 8; ++r) rinv[r] = __builtin_amdgcn_rcpf(lvec[r]);
#pragma unroll
    for (int j = 0; j < 4; ++j)
#pragma unroll
        for (int r = 0; r < 8; ++r)
            out[(size_t)(b * NN + qt * 16 + hi * 8 + r) * GG + j * 16 + col] = O[j][r] * rinv[r];
}

extern "C" void kernel_launch(void* const* d_in, const int* in_sizes, int n_in,
                              void* d_out, int out_size, void* d_ws, size_t ws_size,
                              hipStream_t stream) {
    (void)in_sizes; (void)n_in; (void)out_size; (void)ws_size;
    const float* x    = (const float*)d_in[0];
    const int*   vlen = (const int*)  d_in[1];
    const float* Wq   = (const float*)d_in[2];
    const float* bq   = (const float*)d_in[3];
    const float* Wk   = (const float*)d_in[4];
    const float* bk   = (const float*)d_in[5];
    const float* Wv   = (const float*)d_in[6];
    const float* bv   = (const float*)d_in[7];
    float* out = (float*)d_out;

    __bf16* wsb = (__bf16*)d_ws;
    __bf16* wt  = wsb;                           // 3 * 64*64 bf16
    __bf16* Qb  = wsb + 3 * CC * GG;             // B*N*G bf16
    __bf16* Kb  = Qb + (size_t)BB * NN * GG;     // B*N*G bf16
    __bf16* Vt  = Kb + (size_t)BB * NN * GG;     // B*G*N bf16 (transposed)

    sattn_wt_kernel<<<3, 256, 0, stream>>>(Wq, Wk, Wv, wt);
    sattn_qkv_kernel<<<BB * (NN / 16), 32, 0, stream>>>(x, wt, bq, bk, bv, Qb, Kb, Vt);
    sattn_attn_kernel<<<BB * (NN / 16), 32, 0, stream>>>(Qb, Kb, Vt, vlen, out);
}
